// Attention_11046655885475
// MI455X (gfx1250) — matrix operand order repaired and checked by reading, not compiled, not run on hardware
//
#include <hip/hip_runtime.h>
#include <stdint.h>

#define HEADS 16
#define DIMH  64
#define NB    2
#define CCH   1024
#define NSEQ  2048
#define HDIM  (HEADS * DIMH)   // 1024

typedef __attribute__((ext_vector_type(16))) __bf16   v16bf;
typedef __attribute__((ext_vector_type(8)))  __bf16   v8bf;
typedef __attribute__((ext_vector_type(8)))  float    v8f;
typedef __attribute__((ext_vector_type(4)))  uint32_t v4u;
typedef __attribute__((ext_vector_type(8)))  uint32_t v8u;

union frag16 { v16bf v; v8bf h[2]; };

static __device__ __forceinline__ v8f wmma_bf16(const frag16& a, const frag16& b, v8f c) {
  // D = A(16x32 bf16) x B(32x16 bf16) + C(16x16 f32)
  return __builtin_amdgcn_wmma_f32_16x16x32_bf16(false, a.v, false, b.v,
                                                 (short)0, c, false, false);
}

// Low 32 bits of a flat pointer into the LDS aperture == LDS byte offset (ISA 10.2).
static __device__ __forceinline__ uint32_t lds_off(const void* p) {
  return (uint32_t)(uintptr_t)p;
}

// TDM: 2-D tile load, bf16 elements. dims/strides in elements (data_size units).
// D# group0/group1 packed per CDNA5 ISA 8.3/8.4; 2-group form (<=2D tensor).
static __device__ __forceinline__ void tdm_load_2d(
    uint32_t lds_addr, const void* gptr,
    uint32_t dim0, uint32_t dim1, uint32_t tile0, uint32_t tile1, uint32_t stride0)
{
  const uint64_t ga = (uint64_t)(uintptr_t)gptr;
  v4u g0;
  g0[0] = 1u;                                                  // count=1, user mode
  g0[1] = lds_addr;                                            // lds_addr[31:0]
  g0[2] = (uint32_t)ga;                                        // global_addr[31:0]
  g0[3] = (uint32_t)((ga >> 32) & 0x01FFFFFFu) | (2u << 30);   // addr[56:32] | type=2
  v8u g1;
  g1[0] = (1u << 16);                                          // data_size=1 (2 bytes)
  g1[1] = (dim0 & 0xFFFFu) << 16;                              // tensor_dim0[15:0]
  g1[2] = (dim0 >> 16) | ((dim1 & 0xFFFFu) << 16);             // dim0[31:16] | dim1[15:0]
  g1[3] = (dim1 >> 16) | (tile0 << 16);                        // dim1[31:16] | tile_dim0
  g1[4] = tile1;                                               // tile_dim1 (tile_dim2=0)
  g1[5] = stride0;                                             // tensor_dim0_stride[31:0]
  g1[6] = 0u;                                                  // stride0[47:32]=0, stride1 lo=0
  g1[7] = 0u;                                                  // stride1 hi=0
  asm volatile("tensor_load_to_lds %0, %1" :: "s"(g0), "s"(g1) : "memory");
}

// ---------------------------------------------------------------- converters
__global__ void cvt_bf16(const float* __restrict__ s, __bf16* __restrict__ d, int n) {
  int i = blockIdx.x * 256 + threadIdx.x;
  if (i < n) d[i] = (__bf16)s[i];
}

// x[b][c][n] f32 -> xT[b][n][c] bf16 (K-contiguous for WMMA B fragments)
__global__ void transpose_cvt(const float* __restrict__ x, __bf16* __restrict__ xT) {
  __shared__ __bf16 tile[32][33];
  const int b  = blockIdx.z;
  const int n0 = blockIdx.x * 32, c0 = blockIdx.y * 32;
  const int tx = threadIdx.x, ty = threadIdx.y;   // (32, 8)
  const float* src = x + ((size_t)b * CCH + c0) * NSEQ + n0;
#pragma unroll
  for (int i = 0; i < 4; ++i)
    tile[ty + 8 * i][tx] = (__bf16)src[(size_t)(ty + 8 * i) * NSEQ + tx];
  __syncthreads();
  __bf16* dst = xT + ((size_t)b * NSEQ + n0) * CCH + c0;
#pragma unroll
  for (int i = 0; i < 4; ++i)
    dst[(size_t)(ty + 8 * i) * CCH + tx] = tile[tx][ty + 8 * i];
}

// ---------------------------------------------------------------- GEMM core
// acc[2][4] : 32(M) x 64(N) wave tile.  A: [M][K] row-major bf16.
// Bt: [N][K] row-major bf16 (B stored K-contiguous per output column).
static __device__ __forceinline__ void gemm_tile_32x64(
    const __bf16* __restrict__ A, const __bf16* __restrict__ Bt,
    int K, int m0, int n0, v8f acc[2][4])
{
  const int lo = threadIdx.x & 15;
  const int hi = (threadIdx.x >> 4) & 1;
  const __bf16* arow0 = A + (size_t)(m0 + lo) * K;
  const __bf16* arow1 = A + (size_t)(m0 + 16 + lo) * K;
  const __bf16* brow[4];
#pragma unroll
  for (int ns = 0; ns < 4; ++ns) brow[ns] = Bt + (size_t)(n0 + ns * 16 + lo) * K;

  for (int k0 = 0; k0 < K; k0 += 32) {
    frag16 a0, a1, b[4];
    a0.h[0] = *(const v8bf*)(arow0 + k0 + hi * 8);
    a0.h[1] = *(const v8bf*)(arow0 + k0 + 16 + hi * 8);
    a1.h[0] = *(const v8bf*)(arow1 + k0 + hi * 8);
    a1.h[1] = *(const v8bf*)(arow1 + k0 + 16 + hi * 8);
#pragma unroll
    for (int ns = 0; ns < 4; ++ns) {
      const __bf16* p = brow[ns] + k0 + hi * 8;
      b[ns].h[0] = *(const v8bf*)(p);
      b[ns].h[1] = *(const v8bf*)(p + 16);
    }
    if (k0 + 128 < K) {                 // global_prefetch_b8
      __builtin_prefetch(arow0 + k0 + 128, 0, 1);
      __builtin_prefetch(brow[0] + k0 + 128, 0, 1);
      __builtin_prefetch(brow[2] + k0 + 128, 0, 1);
    }
#pragma unroll
    for (int ns = 0; ns < 4; ++ns) {
      acc[0][ns] = wmma_bf16(a0, b[ns], acc[0][ns]);
      acc[1][ns] = wmma_bf16(a1, b[ns], acc[1][ns]);
    }
  }
}

// qkv[o][n] = sum_c w[o][c] * x[b][c][n]; scatter into Q/K ([B,H,N,D]) and V ([B,H,D,N])
__global__ void __launch_bounds__(256) qkv_gemm(
    const __bf16* __restrict__ W, const __bf16* __restrict__ xT,
    __bf16* __restrict__ Q, __bf16* __restrict__ Km, __bf16* __restrict__ Vm)
{
  const int wid = threadIdx.x >> 5;
  const int b   = blockIdx.z;
  const int m0  = blockIdx.y * 64 + (wid >> 2) * 32;
  const int n0  = blockIdx.x * 256 + (wid & 3) * 64;
  v8f acc[2][4] = {};
  gemm_tile_32x64(W, xT + (size_t)b * NSEQ * CCH, CCH, m0, n0, acc);

  const int lo = threadIdx.x & 15, hi = (threadIdx.x >> 4) & 1;
  const float scale = 0.125f;           // DIM_HEAD^-0.5
#pragma unroll
  for (int ms = 0; ms < 2; ++ms)
#pragma unroll
    for (int ns = 0; ns < 4; ++ns)
#pragma unroll
      for (int r = 0; r < 8; ++r) {
        const int o = m0 + ms * 16 + r + 8 * hi;
        const int n = n0 + ns * 16 + lo;
        const float v = acc[ms][ns][r];
        const int seg = o >> 10, oo = o & 1023, h = oo >> 6, d = oo & 63;
        const size_t bh = (size_t)b * HEADS + h;
        if (seg == 0)      Q [(bh * NSEQ + n) * DIMH + d] = (__bf16)(v * scale);
        else if (seg == 1) Km[(bh * NSEQ + n) * DIMH + d] = (__bf16)v;
        else               Vm[(bh * DIMH + d) * NSEQ + n] = (__bf16)v;
      }
}

// ------------------------------------------------------------ flash attention
// Block = 4 waves on 4 consecutive 16-query tiles of the same (b,h).
// K/V tiles (64 KV steps of 32 keys) staged in LDS by wave 0 via TDM, double-buffered,
// tracked with TENSORcnt, published via workgroup split-barrier.
__global__ void __launch_bounds__(128) flash_attn(
    const __bf16* __restrict__ Q, const __bf16* __restrict__ Km,
    const __bf16* __restrict__ Vm, __bf16* __restrict__ O)
{
  __shared__ __bf16 kbuf[2][32 * DIMH];   // [j=32][d=64] per buffer (4 KB)
  __shared__ __bf16 vbuf[2][DIMH * 32];   // [d=64][j=32] per buffer (4 KB)
  __shared__ __bf16 pbuf[4][16 * 32];     // per-wave P bounce (C->A layout)

  const int lane = threadIdx.x & 31;
  const int wid  = threadIdx.x >> 5;
  const int lo = lane & 15, hi = lane >> 4;
  const int bh  = blockIdx.x >> 5;                 // 32 blocks per (b,h)
  const int qt  = ((blockIdx.x & 31) << 2) + wid;  // query tile of this wave
  const int qi0 = qt * 16;

  const __bf16* Qb = Q  + ((size_t)bh * NSEQ + qi0) * DIMH;
  const __bf16* Kb = Km + (size_t)bh * NSEQ * DIMH;
  const __bf16* Vb = Vm + (size_t)bh * DIMH * NSEQ;

  frag16 aq[2];
#pragma unroll
  for (int kd = 0; kd < 2; ++kd) {
    const __bf16* p = Qb + (size_t)lo * DIMH + kd * 32 + hi * 8;
    aq[kd].h[0] = *(const v8bf*)(p);
    aq[kd].h[1] = *(const v8bf*)(p + 16);
  }

  v8f acc[4] = {};
  float rmax[8], rsum[8];
#pragma unroll
  for (int r = 0; r < 8; ++r) { rmax[r] = -1e30f; rsum[r] = 0.f; }

  __bf16* pb = pbuf[wid];

  if (wid == 0) {   // preload KV step 0
    tdm_load_2d(lds_off(kbuf[0]), Kb, DIMH, NSEQ, DIMH, 32, DIMH);
    tdm_load_2d(lds_off(vbuf[0]), Vb, NSEQ, DIMH, 32, DIMH, NSEQ);
  }

  const int NT = NSEQ / 32;
  for (int jt = 0; jt < NT; ++jt) {
    const int cur = jt & 1;
    if (wid == 0) {
      if (jt + 1 < NT) {      // issue next tile ahead, then wait for current
        const int j1 = (jt + 1) * 32;
        tdm_load_2d(lds_off(kbuf[cur ^ 1]), Kb + (size_t)j1 * DIMH,
                    DIMH, NSEQ, DIMH, 32, DIMH);
        tdm_load_2d(lds_off(vbuf[cur ^ 1]), Vb + j1,
                    NSEQ, DIMH, 32, DIMH, NSEQ);
        __builtin_amdgcn_s_wait_tensorcnt(2);
      } else {
        __builtin_amdgcn_s_wait_tensorcnt(0);
      }
    }
    __syncthreads();          // tile jt visible to all 4 waves

    const __bf16* Kt = kbuf[cur];
    const __bf16* Vt = vbuf[cur];

    v8f sim[2];
#pragma unroll
    for (int jh = 0; jh < 2; ++jh) {
      frag16 bk0, bk1;
      const __bf16* p = Kt + (size_t)(jh * 16 + lo) * DIMH + hi * 8;
      bk0.h[0] = *(const v8bf*)(p);          bk0.h[1] = *(const v8bf*)(p + 16);
      bk1.h[0] = *(const v8bf*)(p + 32);     bk1.h[1] = *(const v8bf*)(p + 48);
      v8f c = {};
      c = wmma_bf16(aq[0], bk0, c);
      c = wmma_bf16(aq[1], bk1, c);
      sim[jh] = c;
    }

#pragma unroll
    for (int r = 0; r < 8; ++r) {
      float nm = fmaxf(rmax[r], fmaxf(sim[0][r], sim[1][r]));
      nm = fmaxf(nm, __shfl_xor(nm, 1, 32));
      nm = fmaxf(nm, __shfl_xor(nm, 2, 32));
      nm = fmaxf(nm, __shfl_xor(nm, 4, 32));
      nm = fmaxf(nm, __shfl_xor(nm, 8, 32));
      const float corr = __expf(rmax[r] - nm);
      rmax[r] = nm;
      const float p0 = __expf(sim[0][r] - nm);
      const float p1 = __expf(sim[1][r] - nm);
      float s = p0 + p1;
      s += __shfl_xor(s, 1, 32);
      s += __shfl_xor(s, 2, 32);
      s += __shfl_xor(s, 4, 32);
      s += __shfl_xor(s, 8, 32);
      rsum[r] = rsum[r] * corr + s;
#pragma unroll
      for (int dt = 0; dt < 4; ++dt) acc[dt][r] = acc[dt][r] * corr;
      const int m = r + 8 * hi;
      pb[m * 32 + lo]      = (__bf16)p0;
      pb[m * 32 + 16 + lo] = (__bf16)p1;
    }

    __builtin_amdgcn_wave_barrier();
    asm volatile("s_wait_dscnt 0x0" ::: "memory");   // P tile RAW (DS in-order per wave)

    frag16 ap;
    {
      const __bf16* pr = pb + lo * 32 + hi * 8;
      ap.h[0] = *(const v8bf*)(pr);
      ap.h[1] = *(const v8bf*)(pr + 16);
    }
    __builtin_amdgcn_wave_barrier();

#pragma unroll
    for (int dt = 0; dt < 4; ++dt) {
      frag16 bv;
      const __bf16* p = Vt + (size_t)(dt * 16 + lo) * 32 + hi * 8;
      bv.h[0] = *(const v8bf*)(p);
      bv.h[1] = *(const v8bf*)(p + 16);
      acc[dt] = wmma_bf16(ap, bv, acc[dt]);
    }

    asm volatile("s_wait_dscnt 0x0" ::: "memory");   // DS reads done before producer reuses buf
    __syncthreads();
  }

  // epilogue: O[b][n][h*64+d] so the projection's B operand is K-contiguous
  const int b = bh >> 4, h = bh & 15;
  __bf16* Ob = O + ((size_t)b * NSEQ + qi0) * HDIM + h * DIMH;
#pragma unroll
  for (int r = 0; r < 8; ++r) {
    const float inv = 1.f / rsum[r];
    const int m = r + 8 * hi;
#pragma unroll
    for (int dt = 0; dt < 4; ++dt)
      Ob[(size_t)m * HDIM + dt * 16 + lo] = (__bf16)(acc[dt][r] * inv);
  }
}

// out[b][o][n] = sum_hd w_out[o][hd] * O[b][n][hd] + bias[o]   (f32 output)
__global__ void __launch_bounds__(256) proj_gemm(
    const __bf16* __restrict__ W, const __bf16* __restrict__ Ot,
    const float* __restrict__ bias, float* __restrict__ out)
{
  const int wid = threadIdx.x >> 5;
  const int b   = blockIdx.z;
  const int m0  = blockIdx.y * 64 + (wid >> 2) * 32;
  const int n0  = blockIdx.x * 256 + (wid & 3) * 64;
  v8f acc[2][4] = {};
  gemm_tile_32x64(W, Ot + (size_t)b * NSEQ * HDIM, HDIM, m0, n0, acc);

  const int lo = threadIdx.x & 15, hi = (threadIdx.x >> 4) & 1;
#pragma unroll
  for (int ms = 0; ms < 2; ++ms)
#pragma unroll
    for (int ns = 0; ns < 4; ++ns)
#pragma unroll
      for (int r = 0; r < 8; ++r) {
        const int o = m0 + ms * 16 + r + 8 * hi;
        const int n = n0 + ns * 16 + lo;
        out[((size_t)b * CCH + o) * NSEQ + n] = acc[ms][ns][r] + bias[o];
      }
}

// ---------------------------------------------------------------- launcher
extern "C" void kernel_launch(void* const* d_in, const int* in_sizes, int n_in,
                              void* d_out, int out_size, void* d_ws, size_t ws_size,
                              hipStream_t stream) {
  const float* x     = (const float*)d_in[0];
  const float* w_qkv = (const float*)d_in[1];
  const float* w_out = (const float*)d_in[2];
  const float* b_out = (const float*)d_in[3];
  float* out = (float*)d_out;

  char* ws = (char*)d_ws;                       // 48 MB total layout
  __bf16* xT = (__bf16*)(ws);                                  //  8 MB  [B][N][C]
  __bf16* Wq = (__bf16*)(ws + ((size_t)8  << 20));             //  6 MB  [3072][1024]
  __bf16* Wo = (__bf16*)(ws + ((size_t)14 << 20));             //  2 MB  [1024][1024]
  __bf16* Q  = (__bf16*)(ws + ((size_t)16 << 20));             //  8 MB  [B,H,N,D]
  __bf16* Km = (__bf16*)(ws + ((size_t)24 << 20));             //  8 MB  [B,H,N,D]
  __bf16* Vm = (__bf16*)(ws + ((size_t)32 << 20));             //  8 MB  [B,H,D,N]
  __bf16* O  = (__bf16*)(ws + ((size_t)40 << 20));             //  8 MB  [B][N][HD]

  cvt_bf16<<<(3 * HDIM * CCH + 255) / 256, 256, 0, stream>>>(w_qkv, Wq, 3 * HDIM * CCH);
  cvt_bf16<<<(CCH * HDIM + 255) / 256, 256, 0, stream>>>(w_out, Wo, CCH * HDIM);
  transpose_cvt<<<dim3(NSEQ / 32, CCH / 32, NB), dim3(32, 8), 0, stream>>>(x, xT);
  qkv_gemm<<<dim3(NSEQ / 256, 3 * HDIM / 64, NB), 256, 0, stream>>>(Wq, xT, Q, Km, Vm);
  flash_attn<<<dim3(NB * HEADS * (NSEQ / 16) / 4), 128, 0, stream>>>(Q, Km, Vm, O);
  proj_gemm<<<dim3(NSEQ / 256, CCH / 64, NB), 256, 0, stream>>>(Wo, O, b_out, out);
}
